// S5TransitionModel_30305289240740
// MI455X (gfx1250) — compile-verified, matrix operand order checked by reading, not run on hardware
//
#include <hip/hip_runtime.h>
#include <hip/hip_bf16.h>

// ---------------------------------------------------------------------------
// S5 transition model, fused CDNA5 (gfx1250) implementation.
// B=8192, LATENT=256, D=128, U=32, NY=40, HID=128.
// GEMMs: v_wmma_f32_16x16x32_f16. Data movement: global_load_async_to_lds_b128
// (ASYNCcnt) for activation tiles, TDM tensor_load_to_lds (TENSORcnt) for
// weight-chunk streaming in the fused head kernels.
// ---------------------------------------------------------------------------

#define B_SZ   8192
#define LATENT 256
#define DDIM   128
#define UDIM   32
#define NYN    40
#define HIDN   128
#define KIN    289
#define KINP   320   // 289 padded to multiple of 32
#define BN_EPS 1e-5f

#define USE_ASYNC_LDS 1
#define USE_TDM       1

typedef __attribute__((ext_vector_type(16))) _Float16 v16h;
typedef __attribute__((ext_vector_type(8)))  _Float16 v8h;
typedef __attribute__((ext_vector_type(8)))  float    v8f;
typedef unsigned int u32;
typedef u32 u32x4 __attribute__((ext_vector_type(4)));
typedef u32 u32x8 __attribute__((ext_vector_type(8)));

__device__ __forceinline__ u32 lds_off32(const void* p) {
  return (u32)(uintptr_t)p;   // generic LDS pointer: addr[31:0] = LDS offset
}

#if USE_ASYNC_LDS
// Async global -> LDS copy (16B), tracked with ASYNCcnt.
__device__ __forceinline__ void async_ld_b128(u32 ldsoff, const void* g) {
  asm volatile("global_load_async_to_lds_b128 %0, %1, off"
               :: "v"(ldsoff), "v"((unsigned long long)(uintptr_t)g)
               : "memory");
}
__device__ __forceinline__ void wait_async0() {
  asm volatile("s_wait_asynccnt 0x0" ::: "memory");
}
#endif

#if USE_TDM
// TDM: DMA a 2D tile (tile_dim0=32 halves x tile_dim1=128 rows) from a
// row-major [rows,256] f16 tensor into contiguous LDS. D# per ISA ch8.
__device__ __forceinline__ void tdm_load_tile_32x128(u32 ldsoff, const void* gsrc) {
  const unsigned long long ga = (unsigned long long)(uintptr_t)gsrc;
  u32x4 g0;
  g0[0] = 1u;                                          // count=1 (user D#)
  g0[1] = ldsoff;                                      // lds_addr
  g0[2] = (u32)ga;                                     // global_addr[31:0]
  g0[3] = (u32)((ga >> 32) & 0x01FFFFFFu) | (2u << 30); // addr[56:32] | type=2
  u32x8 g1;
  g1[0] = (1u << 16);        // workgroup_mask=0 | data_size=1 (2 bytes)
  g1[1] = (256u << 16);      // atomic_barrier_addr=0 | tensor_dim0[15:0]=256
  g1[2] = (128u << 16);      // tensor_dim0[31:16]=0 | tensor_dim1[15:0]=128
  g1[3] = (32u << 16);       // tensor_dim1[31:16]=0 | tile_dim0=32
  g1[4] = 128u;              // tile_dim1=128 | tile_dim2=0
  g1[5] = 256u;              // tensor_dim0_stride[31:0]=256 (elements)
  g1[6] = 0u;                // stride hi | tensor_dim1_stride lo
  g1[7] = 0u;
  asm volatile("tensor_load_to_lds %0, %1" :: "s"(g0), "s"(g1) : "memory");
}
#endif

// ---- WMMA fragment loader --------------------------------------------------
// A (16x32 MxK) and B (32x16 KxN as W[N][K] row-major) per ISA 7.12.2:
// lane r in [0,16): row r, K chunks {hi*8..hi*8+7} and {16+hi*8..}, hi=lane>>4.
__device__ __forceinline__ v16h load_frag_ab(const _Float16* base, int row_stride) {
  const int lane = threadIdx.x & 31;
  const int r    = lane & 15;
  const int hi   = lane >> 4;
  const _Float16* p = base + (size_t)r * row_stride + hi * 8;
  v8h lo = *(const v8h*)(p);
  v8h hs = *(const v8h*)(p + 16);
  v16h f;
#pragma unroll
  for (int i = 0; i < 8; ++i) { f[i] = lo[i]; f[i + 8] = hs[i]; }
  return f;
}

__device__ __forceinline__ void zero_acc(v8f* acc) {
#pragma unroll
  for (int m = 0; m < 8; ++m)
#pragma unroll
    for (int i = 0; i < 8; ++i) acc[m][i] = 0.0f;
}

#define WMMA16(a, b, c) \
  __builtin_amdgcn_wmma_f32_16x16x32_f16(false, (a), false, (b), (short)0, (c), false, false)

// ---------------------------------------------------------------------------
// Generic 128x128 tile GEMM: C[128,128] = A[128,K] * W[N,K]^T (+bias)
// amode: 0 = A f16 (async->LDS), 1 = A f32 (convert), 2 = f32 with per-k
//        scale/shift + ReLU (fused BatchNorm+ReLU). cmode: 0 f32, 1 f16.
// grid = (B/128, N/128), block = 256 (8 waves). dyn LDS = 2*128*32 halves.
// ---------------------------------------------------------------------------
__global__ void k_gemm(const void* Aptr, int lda, int amode,
                       const float* ascale, const float* ashift,
                       const _Float16* Bw, const float* bias,
                       void* Cptr, int ldc, int cmode, int K) {
  extern __shared__ char smem_raw[];
  _Float16* ldsA = (_Float16*)smem_raw;   // 128 x 32
  _Float16* ldsB = ldsA + 128 * 32;       // 128 x 32
  const int row0 = blockIdx.x * 128;
  const int n0   = blockIdx.y * 128;
  const int t    = threadIdx.x;
  const int wave = t >> 5;
  const int lane = t & 31;

  v8f acc[8];
  zero_acc(acc);

  const int nk = K >> 5;
  const int ar = t >> 1;
  const int as = (t & 1) << 4;
#pragma unroll 1
  for (int kk = 0; kk < nk; ++kk) {
    const int k0 = kk << 5;
    // ---- stage A chunk (128 x 32) -----------------------------------------
    {
      _Float16* dst = ldsA + ar * 32 + as;
      if (amode == 0) {
        const _Float16* src = (const _Float16*)Aptr + (size_t)(row0 + ar) * lda + k0 + as;
#if USE_ASYNC_LDS
        const u32 doff = lds_off32(dst);
        async_ld_b128(doff, src);
        async_ld_b128(doff + 16, src + 8);
#else
        *(v8h*)dst       = *(const v8h*)src;
        *(v8h*)(dst + 8) = *(const v8h*)(src + 8);
#endif
      } else if (amode == 1) {
        const float* src = (const float*)Aptr + (size_t)(row0 + ar) * lda + k0 + as;
#pragma unroll
        for (int i = 0; i < 16; ++i) dst[i] = (_Float16)src[i];
      } else {
        const float* src = (const float*)Aptr + (size_t)(row0 + ar) * lda + k0 + as;
#pragma unroll
        for (int i = 0; i < 16; ++i) {
          float v = fmaxf(src[i] * ascale[k0 + as + i] + ashift[k0 + as + i], 0.0f);
          dst[i] = (_Float16)v;
        }
      }
      // ---- stage B chunk: weight rows n0..n0+127, k chunk -----------------
      const _Float16* wsrc = Bw + (size_t)(n0 + ar) * K + k0 + as;
      _Float16* wdst = ldsB + ar * 32 + as;
#if USE_ASYNC_LDS
      const u32 woff = lds_off32(wdst);
      async_ld_b128(woff, wsrc);
      async_ld_b128(woff + 16, wsrc + 8);
#else
      *(v8h*)wdst       = *(const v8h*)wsrc;
      *(v8h*)(wdst + 8) = *(const v8h*)(wsrc + 8);
#endif
      if (kk + 1 < nk)  // L2-hot weight stream hint -> global_prefetch_b8
        __builtin_prefetch(wsrc + 32, 0, 1);
    }
#if USE_ASYNC_LDS
    wait_async0();
#endif
    __syncthreads();
    v16h bfrag = load_frag_ab(ldsB + wave * 16 * 32, 32);
#pragma unroll
    for (int m = 0; m < 8; ++m) {
      v16h afrag = load_frag_ab(ldsA + m * 16 * 32, 32);
      acc[m] = WMMA16(afrag, bfrag, acc[m]);
    }
    __syncthreads();
  }

  // ---- epilogue: C layout lane = (hi, n), vgpr i -> M = i + 8*hi ----------
  const int hi  = lane >> 4;
  const int col = n0 + wave * 16 + (lane & 15);
  const float bb = bias ? bias[col] : 0.0f;
#pragma unroll
  for (int m = 0; m < 8; ++m) {
#pragma unroll
    for (int i = 0; i < 8; ++i) {
      const int row = row0 + m * 16 + i + hi * 8;
      const float v = acc[m][i] + bb;
      if (cmode == 0) ((float*)Cptr)[(size_t)row * ldc + col] = v;
      else            ((_Float16*)Cptr)[(size_t)row * ldc + col] = (_Float16)v;
    }
  }
}

// ---------------------------------------------------------------------------
// Fused-head helper: 128x128 chunk GEMM over K=256 with the full A tile
// (128x256 f16) resident in LDS. Weight chunks streamed with TDM, double-
// buffered across two 8 KB LDS tiles (DMA of k+1 overlaps WMMA of k).
// ---------------------------------------------------------------------------
__device__ __forceinline__ void gemm_chunk256(const _Float16* ldsA, const _Float16* Wrows,
                                              _Float16* ldsB0, _Float16* ldsB1, v8f* acc) {
  const int t = threadIdx.x, wave = t >> 5;
#if USE_TDM
  if (wave == 0) tdm_load_tile_32x128(lds_off32(ldsB0), Wrows);
#pragma unroll 1
  for (int kk = 0; kk < 8; ++kk) {
    const _Float16* cur = (kk & 1) ? ldsB1 : ldsB0;
    if (wave == 0) {
      if (kk < 7) {
        tdm_load_tile_32x128(lds_off32((kk & 1) ? ldsB0 : ldsB1),
                             Wrows + ((kk + 1) << 5));
        __builtin_amdgcn_s_wait_tensorcnt((short)1);
      } else {
        __builtin_amdgcn_s_wait_tensorcnt((short)0);
      }
    }
    __syncthreads();
    v16h bfrag = load_frag_ab(cur + wave * 16 * 32, 32);
#pragma unroll
    for (int m = 0; m < 8; ++m) {
      v16h afrag = load_frag_ab(ldsA + (size_t)(m * 16) * 256 + (kk << 5), 256);
      acc[m] = WMMA16(afrag, bfrag, acc[m]);
    }
    __syncthreads();
  }
#else
  const int ar = t >> 1, as = (t & 1) << 4;
#pragma unroll 1
  for (int kk = 0; kk < 8; ++kk) {
    const _Float16* wsrc = Wrows + (size_t)ar * 256 + (kk << 5) + as;
    _Float16* wdst = ldsB0 + ar * 32 + as;
    *(v8h*)wdst       = *(const v8h*)wsrc;
    *(v8h*)(wdst + 8) = *(const v8h*)(wsrc + 8);
    __syncthreads();
    v16h bfrag = load_frag_ab(ldsB0 + wave * 16 * 32, 32);
#pragma unroll
    for (int m = 0; m < 8; ++m) {
      v16h afrag = load_frag_ab(ldsA + (size_t)(m * 16) * 256 + (kk << 5), 256);
      acc[m] = WMMA16(afrag, bfrag, acc[m]);
    }
    __syncthreads();
  }
  (void)ldsB1;
#endif
}

__device__ __forceinline__ void stage_hz_tile(const _Float16* hz, int row0, _Float16* ldsA) {
  const int t = threadIdx.x;
  const int ar = t >> 1, cs = (t & 1) * 128;
  const _Float16* src = hz + (size_t)(row0 + ar) * 256 + cs;
  _Float16* dst = ldsA + (size_t)ar * 256 + cs;
#if USE_ASYNC_LDS
  const u32 doff = lds_off32(dst);
#pragma unroll
  for (int i = 0; i < 128; i += 8) async_ld_b128(doff + i * 2, src + i);
  wait_async0();
#else
#pragma unroll
  for (int i = 0; i < 128; i += 8) *(v8h*)(dst + i) = *(const v8h*)(src + i);
#endif
}

// ---------------------------------------------------------------------------
// k_bt: z_next = z_rot + einsum('bdu,bu->bd', Bt(hz), ut*dt)   (complex)
// wbt packed [8192,256]: rows 0..4095 = Bt_r_W, 4096..8191 = Bt_i_W.
// 64 N-chunks of 128 cols; chunk c covers d in [4c,4c+4) (all 32 u's).
// Writes z_next (re|im per row) straight into d_out[:, 0:256].
// ---------------------------------------------------------------------------
__global__ void k_bt(const _Float16* hz, const _Float16* wbt,
                     const float* bt_r_b, const float* bt_i_b,
                     const float* utdt, const float* zrot, float* outz) {
  extern __shared__ char smem_raw[];
  _Float16* ldsA  = (_Float16*)smem_raw;           // 128*256 f16 = 64KB
  _Float16* ldsB0 = ldsA + 128 * 256;              // 128*32  f16 =  8KB
  _Float16* ldsB1 = ldsB0 + 128 * 32;              // 128*32  f16 =  8KB
  float*    ldsC  = (float*)(ldsB1 + 128 * 32);    // 128*128 f32 = 64KB
  const int row0 = blockIdx.x * 128;
  const int t = threadIdx.x, wave = t >> 5, lane = t & 31;

  stage_hz_tile(hz, row0, ldsA);
  __syncthreads();

#pragma unroll 1
  for (int c = 0; c < 64; ++c) {
    v8f acc[8];
    zero_acc(acc);
    gemm_chunk256(ldsA, wbt + (size_t)c * 128 * 256, ldsB0, ldsB1, acc);

    const int hi = lane >> 4;
    const int nl = wave * 16 + (lane & 15);       // local col 0..127
    const int gn = c * 128 + nl;                  // flat 0..8191
    const float bb = (c < 32) ? bt_r_b[gn] : bt_i_b[gn - 4096];
#pragma unroll
    for (int m = 0; m < 8; ++m)
#pragma unroll
      for (int i = 0; i < 8; ++i)
        ldsC[(size_t)(m * 16 + i + hi * 8) * 128 + nl] = acc[m][i] + bb;
    __syncthreads();

    const bool isI = (c >= 32);
    const int  cc  = isI ? c - 32 : c;
    for (int o = t; o < 512; o += 256) {
      const int rl = o >> 2, dl = o & 3;
      const float* u  = utdt + (size_t)(row0 + rl) * UDIM;
      const float* cw = ldsC + (size_t)rl * 128 + dl * 32;
      float s = 0.0f;
#pragma unroll
      for (int uu = 0; uu < 32; ++uu) s += cw[uu] * u[uu];
      const int ocol = (isI ? 128 : 0) + cc * 4 + dl;
      const size_t oi = (size_t)(row0 + rl) * 256 + ocol;
      outz[oi] = zrot[oi] + s;
    }
    __syncthreads();
  }
}

// ---------------------------------------------------------------------------
// k_ctdt: yt = Re(einsum('bnd,bd->bn', Ct, z_next)) + einsum('bnu,bu->bn', Dt, ut*dt)
// wct packed [10240,256] (Ct_r rows then Ct_i rows); wdt [1280,256].
// 90 chunks: 0..39 Ct_r (chunk c = output n=c, all d), 40..79 Ct_i, 80..89 Dt.
// ---------------------------------------------------------------------------
__global__ void k_ctdt(const _Float16* hz, const _Float16* wct,
                       const float* ct_r_b, const float* ct_i_b,
                       const _Float16* wdt, const float* dt_b,
                       const float* utdt, const float* znext, float* yt) {
  extern __shared__ char smem_raw[];
  _Float16* ldsA  = (_Float16*)smem_raw;           // 64KB
  _Float16* ldsB0 = ldsA + 128 * 256;              //  8KB
  _Float16* ldsB1 = ldsB0 + 128 * 32;              //  8KB
  float*    ldsC  = (float*)(ldsB1 + 128 * 32);    // 64KB
  float*    ldsY  = ldsC + 128 * 128;              // 128*40 f32 = 20KB
  float*    ldsR  = ldsY + 128 * NYN;              // 256 f32
  const int row0 = blockIdx.x * 128;
  const int t = threadIdx.x, wave = t >> 5, lane = t & 31;

  stage_hz_tile(hz, row0, ldsA);
  for (int o = t; o < 128 * NYN; o += 256) ldsY[o] = 0.0f;
  __syncthreads();

#pragma unroll 1
  for (int c = 0; c < 90; ++c) {
    const _Float16* Wrows = (c < 80) ? (wct + (size_t)c * 128 * 256)
                                     : (wdt + (size_t)(c - 80) * 128 * 256);
    v8f acc[8];
    zero_acc(acc);
    gemm_chunk256(ldsA, Wrows, ldsB0, ldsB1, acc);

    const int hi = lane >> 4;
    const int nl = wave * 16 + (lane & 15);
    float bb;
    if      (c < 40) bb = ct_r_b[c * 128 + nl];
    else if (c < 80) bb = ct_i_b[(c - 40) * 128 + nl];
    else             bb = dt_b[(c - 80) * 128 + nl];
#pragma unroll
    for (int m = 0; m < 8; ++m)
#pragma unroll
      for (int i = 0; i < 8; ++i)
        ldsC[(size_t)(m * 16 + i + hi * 8) * 128 + nl] = acc[m][i] + bb;
    __syncthreads();

    if (c < 80) {
      const bool isI = (c >= 40);
      const int  n   = isI ? c - 40 : c;
      const int  rl  = t >> 1, half = t & 1;
      const float* zn = znext + (size_t)(row0 + rl) * 256 + (isI ? 128 : 0) + half * 64;
      const float* cw = ldsC + (size_t)rl * 128 + half * 64;
      float s = 0.0f;
#pragma unroll
      for (int j = 0; j < 64; ++j) s += cw[j] * zn[j];
      ldsR[t] = s;
      __syncthreads();
      if (half == 0) {
        const float v = ldsR[t] + ldsR[t + 1];
        ldsY[(size_t)rl * NYN + n] += isI ? -v : v;
      }
    } else {
      for (int o = t; o < 512; o += 256) {
        const int rl = o >> 2, nlck = o & 3;
        const float* u  = utdt + (size_t)(row0 + rl) * UDIM;
        const float* cw = ldsC + (size_t)rl * 128 + nlck * 32;
        float s = 0.0f;
#pragma unroll
        for (int uu = 0; uu < 32; ++uu) s += cw[uu] * u[uu];
        ldsY[(size_t)rl * NYN + (c - 80) * 4 + nlck] += s;
      }
    }
    __syncthreads();
  }

  for (int o = t; o < 128 * NYN; o += 256) {
    const int rl = o / NYN, n = o % NYN;
    yt[(size_t)(row0 + rl) * NYN + n] = ldsY[o];
  }
}

// ---------------------------------------------------------------------------
// Small utility kernels
// ---------------------------------------------------------------------------
__global__ void k_cvt(const float* src, _Float16* dst, int rows, int scols, int dcols) {
  const long long o = (long long)blockIdx.x * 256 + threadIdx.x;
  if (o >= (long long)rows * dcols) return;
  const int r = (int)(o / dcols), c = (int)(o % dcols);
  dst[o] = (_Float16)((c < scols) ? src[(size_t)r * scols + c] : 0.0f);
}

// pack x=[zt|ut|dt|0pad] (f16), utdt=ut*dt, waob=[alpha_b|omega_b], zero stats
__global__ void k_prep(const float* zt, const float* dt, const float* ut,
                       const float* ab, const float* ob,
                       _Float16* xf16, float* utdt, float* waob, float* stats) {
  const long long o = (long long)blockIdx.x * 256 + threadIdx.x;
  const long long n1 = (long long)B_SZ * KINP;
  const long long n2 = n1 + (long long)B_SZ * UDIM;
  const long long n3 = n2 + 256;
  const long long n4 = n3 + 512;
  if (o < n1) {
    const int b = (int)(o / KINP), c = (int)(o % KINP);
    float v;
    if      (c < LATENT)        v = zt[(size_t)b * LATENT + c];
    else if (c < LATENT + UDIM) v = ut[(size_t)b * UDIM + (c - LATENT)];
    else if (c == KIN - 1)      v = dt[b];
    else                        v = 0.0f;
    xf16[o] = (_Float16)v;
  } else if (o < n2) {
    const long long i = o - n1;
    const int b = (int)(i / UDIM), u = (int)(i % UDIM);
    utdt[i] = ut[(size_t)b * UDIM + u] * dt[b];
  } else if (o < n3) {
    const int j = (int)(o - n2);
    waob[j] = (j < 128) ? ab[j] : ob[j - 128];
  } else if (o < n4) {
    stats[o - n3] = 0.0f;
  }
}

// per-feature batch stats partial reduction (training-mode BN)
__global__ void k_stat(const float* h, float* sum, float* sq) {
  const int tile = blockIdx.x, t = threadIdx.x;
  const int c = t & 127, rb = t >> 7;
  float s = 0.0f, q = 0.0f;
  for (int r = rb; r < 128; r += 2) {
    const float v = h[(size_t)(tile * 128 + r) * 128 + c];
    s += v; q += v * v;
  }
  atomicAdd(&sum[c], s);
  atomicAdd(&sq[c], q);
}

__global__ void k_bnscale(const float* sum, const float* sq, const float* g,
                          const float* be, float* a, float* bsh) {
  const int c = threadIdx.x;
  if (c >= 128) return;
  const float mu  = sum[c] * (1.0f / (float)B_SZ);
  const float var = sq[c] * (1.0f / (float)B_SZ) - mu * mu;
  const float aa  = g[c] * rsqrtf(var + BN_EPS);
  a[c]   = aa;
  bsh[c] = be[c] - mu * aa;
}

// lam = exp(-softplus(a) + i*w) = sigmoid(-a)*(cos w + i sin w)
__global__ void k_lam(const float* ao, float* lam) {
  const long long o = (long long)blockIdx.x * 256 + threadIdx.x;
  if (o >= (long long)B_SZ * DDIM) return;
  const int b = (int)(o >> 7), d = (int)(o & 127);
  const float a = ao[(size_t)b * 256 + d];
  const float w = ao[(size_t)b * 256 + 128 + d];
  const float s = 1.0f / (1.0f + expf(a));
  lam[(size_t)b * 256 + d]       = s * cosf(w);
  lam[(size_t)b * 256 + 128 + d] = s * sinf(w);
}

// a2 = Lam * z_eig (complex elementwise), f16 output in [re|im] layout
__global__ void k_lammul(const float* lam, const float* zeig, _Float16* a2) {
  const long long o = (long long)blockIdx.x * 256 + threadIdx.x;
  if (o >= (long long)B_SZ * DDIM) return;
  const int b = (int)(o >> 7), d = (int)(o & 127);
  const float lr = lam[(size_t)b * 256 + d],  li = lam[(size_t)b * 256 + 128 + d];
  const float er = zeig[(size_t)b * 256 + d], ei = zeig[(size_t)b * 256 + 128 + d];
  a2[(size_t)b * 256 + d]       = (_Float16)(lr * er - li * ei);
  a2[(size_t)b * 256 + 128 + d] = (_Float16)(lr * ei + li * er);
}

// Complex Gauss-Jordan inverse of V (diag-dominant, no pivoting), single WG.
// Packs Wein [256,256] f16 = [[Vinv_r, -Vinv_i],[Vinv_i, Vinv_r]].
__global__ void k_vinv(const float* Vr, const float* Vi, float* aug, _Float16* Wein) {
  __shared__ float fac[128 * 2];
  __shared__ float piv[2];
  const int t = threadIdx.x;
  for (int o = t; o < 128 * 256; o += 256) {
    const int r = o >> 8, c = o & 255;
    float ar, ai;
    if (c < 128) { ar = Vr[(size_t)r * 128 + c]; ai = Vi[(size_t)r * 128 + c]; }
    else         { ar = (c - 128 == r) ? 1.0f : 0.0f; ai = 0.0f; }
    aug[(size_t)o * 2] = ar; aug[(size_t)o * 2 + 1] = ai;
  }
  __syncthreads();
  for (int p = 0; p < 128; ++p) {
    if (t == 0) {
      const float a = aug[((size_t)p * 256 + p) * 2];
      const float b = aug[((size_t)p * 256 + p) * 2 + 1];
      const float d = a * a + b * b;
      piv[0] = a / d; piv[1] = -b / d;
    }
    __syncthreads();
    {
      const int c = t;  // 256 cols, 256 threads
      const size_t i = ((size_t)p * 256 + c) * 2;
      const float xr = aug[i], xi = aug[i + 1];
      aug[i]     = xr * piv[0] - xi * piv[1];
      aug[i + 1] = xr * piv[1] + xi * piv[0];
    }
    __syncthreads();
    if (t < 128) {
      fac[t * 2]     = aug[((size_t)t * 256 + p) * 2];
      fac[t * 2 + 1] = aug[((size_t)t * 256 + p) * 2 + 1];
    }
    __syncthreads();
    for (int o = t; o < 128 * 256; o += 256) {
      const int r = o >> 8, c = o & 255;
      if (r == p) continue;
      const float fr = fac[r * 2], fi = fac[r * 2 + 1];
      const size_t pi2 = ((size_t)p * 256 + c) * 2;
      const float xr = aug[pi2], xi = aug[pi2 + 1];
      aug[(size_t)o * 2]     -= fr * xr - fi * xi;
      aug[(size_t)o * 2 + 1] -= fr * xi + fi * xr;
    }
    __syncthreads();
  }
  for (int o = t; o < 128 * 128; o += 256) {
    const int i = o >> 7, j = o & 127;
    const float vr = aug[((size_t)i * 256 + 128 + j) * 2];
    const float vi = aug[((size_t)i * 256 + 128 + j) * 2 + 1];
    Wein[(size_t)i * 256 + j]               = (_Float16)vr;
    Wein[(size_t)i * 256 + 128 + j]         = (_Float16)(-vi);
    Wein[(size_t)(128 + i) * 256 + j]       = (_Float16)vi;
    Wein[(size_t)(128 + i) * 256 + 128 + j] = (_Float16)vr;
  }
}

// Wrot [256,256] f16 = [[Vr, -Vi],[Vi, Vr]]  (for z_rot = (Lam*z_eig) @ V^T)
__global__ void k_wrot(const float* Vr, const float* Vi, _Float16* Wrot) {
  const long long o = (long long)blockIdx.x * 256 + threadIdx.x;
  if (o >= 128 * 128) return;
  const int i = (int)(o >> 7), j = (int)(o & 127);
  const float vr = Vr[(size_t)i * 128 + j], vi = Vi[(size_t)i * 128 + j];
  Wrot[(size_t)i * 256 + j]               = (_Float16)vr;
  Wrot[(size_t)i * 256 + 128 + j]         = (_Float16)(-vi);
  Wrot[(size_t)(128 + i) * 256 + j]       = (_Float16)vi;
  Wrot[(size_t)(128 + i) * 256 + 128 + j] = (_Float16)vr;
}

// ---------------------------------------------------------------------------
extern "C" void kernel_launch(void* const* d_in, const int* in_sizes, int n_in,
                              void* d_out, int out_size, void* d_ws, size_t ws_size,
                              hipStream_t stream) {
  const float* zt      = (const float*)d_in[0];
  const float* dt      = (const float*)d_in[1];
  const float* ut      = (const float*)d_in[2];
  const float* s0_W    = (const float*)d_in[3];
  const float* s0_g    = (const float*)d_in[5];
  const float* s0_be   = (const float*)d_in[6];
  const float* s1_W    = (const float*)d_in[7];
  const float* s1_g    = (const float*)d_in[9];
  const float* s1_be   = (const float*)d_in[10];
  const float* s2_W    = (const float*)d_in[11];
  const float* s2_b    = (const float*)d_in[12];
  const float* alpha_W = (const float*)d_in[13];
  const float* alpha_b = (const float*)d_in[14];
  const float* omega_W = (const float*)d_in[15];
  const float* omega_b = (const float*)d_in[16];
  const float* Bt_r_W  = (const float*)d_in[17];
  const float* Bt_r_b  = (const float*)d_in[18];
  const float* Bt_i_W  = (const float*)d_in[19];
  const float* Bt_i_b  = (const float*)d_in[20];
  const float* Ct_r_W  = (const float*)d_in[21];
  const float* Ct_r_b  = (const float*)d_in[22];
  const float* Ct_i_W  = (const float*)d_in[23];
  const float* Ct_i_b  = (const float*)d_in[24];
  const float* Dt_W    = (const float*)d_in[25];
  const float* Dt_b    = (const float*)d_in[26];
  const float* V_r     = (const float*)d_in[27];
  const float* V_i     = (const float*)d_in[28];

  float* outz = (float*)d_out;                         // [B,256] z_next (re|im)
  float* outy = (float*)d_out + (size_t)B_SZ * 256;    // [B,40]  yt

  char* ws = (char*)d_ws;
  size_t off = 0;
  auto take = [&](size_t bytes) -> void* {
    void* p = ws + off;
    off = (off + bytes + 255) & ~(size_t)255;
    return p;
  };
  _Float16* xf16  = (_Float16*)take((size_t)B_SZ * KINP * 2);
  _Float16* w0    = (_Float16*)take((size_t)128 * KINP * 2);
  _Float16* w1    = (_Float16*)take((size_t)128 * 128 * 2);
  _Float16* w2    = (_Float16*)take((size_t)256 * 128 * 2);
  _Float16* wao   = (_Float16*)take((size_t)256 * 256 * 2);
  _Float16* wbt   = (_Float16*)take((size_t)8192 * 256 * 2);
  _Float16* wct   = (_Float16*)take((size_t)10240 * 256 * 2);
  _Float16* wdt   = (_Float16*)take((size_t)1280 * 256 * 2);
  float*    waob  = (float*)take(256 * 4);
  float*    stats = (float*)take(512 * 4);   // sum0|sq0|sum1|sq1
  float*    bn0a  = (float*)take(128 * 4);
  float*    bn0b  = (float*)take(128 * 4);
  float*    bn1a  = (float*)take(128 * 4);
  float*    bn1b  = (float*)take(128 * 4);
  float*    h0pre = (float*)take((size_t)B_SZ * 128 * 4);
  float*    h1pre = (float*)take((size_t)B_SZ * 128 * 4);
  _Float16* hzf16 = (_Float16*)take((size_t)B_SZ * 256 * 2);
  float*    aof   = (float*)take((size_t)B_SZ * 256 * 4);
  float*    lam   = (float*)take((size_t)B_SZ * 256 * 4);
  float*    zeig  = (float*)take((size_t)B_SZ * 256 * 4);
  _Float16* a2    = (_Float16*)take((size_t)B_SZ * 256 * 2);
  float*    zrot  = (float*)take((size_t)B_SZ * 256 * 4);
  float*    aug   = (float*)take((size_t)128 * 256 * 2 * 4);
  _Float16* wein  = (_Float16*)take((size_t)256 * 256 * 2);
  _Float16* wrot  = (_Float16*)take((size_t)256 * 256 * 2);
  float*    utdt  = (float*)take((size_t)B_SZ * UDIM * 4);

  auto cvt = [&](const float* s, _Float16* d, int rows, int sc, int dc) {
    const long long n = (long long)rows * dc;
    k_cvt<<<dim3((unsigned)((n + 255) / 256)), dim3(256), 0, stream>>>(s, d, rows, sc, dc);
  };
  // ---- pack weights to f16 ----
  cvt(s0_W, w0, 128, KIN, KINP);
  cvt(s1_W, w1, 128, 128, 128);
  cvt(s2_W, w2, 256, 128, 128);
  cvt(alpha_W, wao, 128, 256, 256);
  cvt(omega_W, wao + (size_t)128 * 256, 128, 256, 256);
  cvt(Bt_r_W, wbt, 4096, 256, 256);
  cvt(Bt_i_W, wbt + (size_t)4096 * 256, 4096, 256, 256);
  cvt(Ct_r_W, wct, 5120, 256, 256);
  cvt(Ct_i_W, wct + (size_t)5120 * 256, 5120, 256, 256);
  cvt(Dt_W, wdt, 1280, 256, 256);

  {
    const long long n = (long long)B_SZ * KINP + (long long)B_SZ * UDIM + 256 + 512;
    k_prep<<<dim3((unsigned)((n + 255) / 256)), dim3(256), 0, stream>>>(
        zt, dt, ut, alpha_b, omega_b, xf16, utdt, waob, stats);
  }
  k_vinv<<<1, 256, 0, stream>>>(V_r, V_i, aug, wein);
  k_wrot<<<dim3((128 * 128 + 255) / 256), dim3(256), 0, stream>>>(V_r, V_i, wrot);

  const size_t gemm_lds = (size_t)2 * 128 * 32 * sizeof(_Float16);  // 16 KB

  // ---- selector ----
  k_gemm<<<dim3(64, 1), 256, gemm_lds, stream>>>(xf16, KINP, 0, nullptr, nullptr,
                                                 w0, nullptr, h0pre, 128, 0, KINP);
  k_stat<<<64, 256, 0, stream>>>(h0pre, stats, stats + 128);
  k_bnscale<<<1, 128, 0, stream>>>(stats, stats + 128, s0_g, s0_be, bn0a, bn0b);
  k_gemm<<<dim3(64, 1), 256, gemm_lds, stream>>>(h0pre, 128, 2, bn0a, bn0b,
                                                 w1, nullptr, h1pre, 128, 0, 128);
  k_stat<<<64, 256, 0, stream>>>(h1pre, stats + 256, stats + 384);
  k_bnscale<<<1, 128, 0, stream>>>(stats + 256, stats + 384, s1_g, s1_be, bn1a, bn1b);
  k_gemm<<<dim3(64, 2), 256, gemm_lds, stream>>>(h1pre, 128, 2, bn1a, bn1b,
                                                 w2, s2_b, hzf16, 256, 1, 128);

  // ---- alpha/omega -> Lam ----
  k_gemm<<<dim3(64, 2), 256, gemm_lds, stream>>>(hzf16, 256, 0, nullptr, nullptr,
                                                 wao, waob, aof, 256, 0, 256);
  k_lam<<<dim3((B_SZ * 128 + 255) / 256), 256, 0, stream>>>(aof, lam);

  // ---- complex eigen-SSM rotation (as real block GEMMs) ----
  k_gemm<<<dim3(64, 2), 256, gemm_lds, stream>>>((const void*)zt, 256, 1, nullptr, nullptr,
                                                 wein, nullptr, zeig, 256, 0, 256);
  k_lammul<<<dim3((B_SZ * 128 + 255) / 256), 256, 0, stream>>>(lam, zeig, a2);
  k_gemm<<<dim3(64, 2), 256, gemm_lds, stream>>>(a2, 256, 0, nullptr, nullptr,
                                                 wrot, nullptr, zrot, 256, 0, 256);

  // ---- fused hypernetwork heads + contractions ----
  const size_t bt_lds = (size_t)128 * 256 * 2 + 2 * (size_t)128 * 32 * 2 +
                        (size_t)128 * 128 * 4;                       // 144 KB
  k_bt<<<64, 256, bt_lds, stream>>>(hzf16, wbt, Bt_r_b, Bt_i_b, utdt, zrot, outz);
  const size_t ct_lds = bt_lds + (size_t)128 * NYN * 4 + 256 * 4;    // ~165 KB
  k_ctdt<<<64, 256, ct_lds, stream>>>(hzf16, wct, Ct_r_b, Ct_i_b, wdt, Dt_b,
                                      utdt, outz, outy);
}
